// MiniKDALayer_43757126811709
// MI455X (gfx1250) — matrix operand order repaired and checked by reading, not compiled, not run on hardware
//
#include <hip/hip_runtime.h>
#include <math.h>

typedef __attribute__((ext_vector_type(16))) __bf16 v16bf;
typedef __attribute__((ext_vector_type(8)))  __bf16 v8bf;
typedef __attribute__((ext_vector_type(8)))  float  v8f;

#define T_SEQ 8192
#define D_IN  1024
#define NCAT  512   // 64 q | 64 k | 64 v | 112 ag | 112 au | 48 d | 48 pad

__device__ __forceinline__ float sigmoidf_(float x){ return 1.0f/(1.0f+expf(-x)); }
__device__ __forceinline__ float siluf_(float x){ return x/(1.0f+expf(-x)); }

// Async DMA: global -> LDS, 16 bytes per lane (CDNA5, tracked by ASYNCcnt).
__device__ __forceinline__ void async_copy_b128(const __bf16* g, __bf16* l) {
    unsigned lofs = (unsigned)(unsigned long long)(void*)l;   // LDS byte offset (low 32 bits)
    asm volatile("global_load_async_to_lds_b128 %0, %1, off"
                 :: "v"(lofs), "v"(g) : "memory");
}

// ---------------------------------------------------------------- weight pack
// All weights packed TRANSPOSED (BT[n][k], K-contiguous) so WMMA B fragments
// are straight b128 loads with no transpose staging.
__global__ void pack_weights_k(const float* __restrict__ Wq, const float* __restrict__ Wk,
                               const float* __restrict__ Wv, const float* __restrict__ Wag,
                               const float* __restrict__ Wau, const float* __restrict__ Wd,
                               const float* __restrict__ Wad, const float* __restrict__ Wu,
                               const float* __restrict__ Wout,
                               __bf16* __restrict__ WcatT, __bf16* __restrict__ WadT,
                               __bf16* __restrict__ WuT, __bf16* __restrict__ WoutT)
{
    int i = blockIdx.x * blockDim.x + threadIdx.x;
    int stride = gridDim.x * blockDim.x;
    for (int idx = i; idx < NCAT*D_IN; idx += stride) {          // WcatT [512][1024]
        int n = idx >> 10, kk = idx & 1023;
        float v = 0.0f;
        if      (n < 64 )  v = Wq[kk*64 + n];
        else if (n < 128)  v = Wk[kk*64 + (n-64)];
        else if (n < 192)  v = Wv[kk*64 + (n-128)];
        else if (n < 304){ int nn = n-192; if (nn < 103) v = Wag[kk*103 + nn]; }
        else if (n < 416){ int nn = n-304; if (nn < 103) v = Wau[kk*103 + nn]; }
        else if (n < 464){ int nn = n-416; if (nn < 39 ) v = Wd [kk*39  + nn]; }
        WcatT[idx] = (__bf16)v;
    }
    for (int idx = i; idx < 128*128; idx += stride) {            // WadT [128][128], K pad 103->128
        int n = idx >> 7, kk = idx & 127;
        WadT[idx] = (__bf16)((kk < 103) ? Wad[kk*128 + n] : 0.0f);
    }
    for (int idx = i; idx < 64*64; idx += stride) {              // WuT [64][64], K pad 39->64
        int n = idx >> 6, kk = idx & 63;
        WuT[idx] = (__bf16)((kk < 39) ? Wu[kk*64 + n] : 0.0f);
    }
    for (int idx = i; idx < 1024*64; idx += stride) {            // WoutT [1024][64]
        int n = idx >> 6, kk = idx & 63;
        WoutT[idx] = (__bf16)Wout[kk*1024 + n];
    }
}

__global__ void convert_x_k(const float* __restrict__ x, __bf16* __restrict__ xb, int n)
{
    int i = blockIdx.x * blockDim.x + threadIdx.x;
    int stride = gridDim.x * blockDim.x;
    for (; i < n; i += stride) xb[i] = (__bf16)x[i];
}

// ---------------------------------------------------------------- WMMA GEMM
// C[M,N] = act(A[M,K] @ B[K,N]); A row-major bf16, B TRANSPOSED (BT[N][K]) bf16,
// C f32. M mult of 256, N mult of 64, K mult of 32.
// Block: 256 thr (8 waves). Block tile 256x64; wave tile 32x64 => 8 WMMAs/K-step
// (A frags reused 4x, B frags reused 2x). Staging via async global->LDS DMA
// (ASYNCcnt), double-buffered: issue next tile, s_wait_asynccnt<=5, barrier.
__global__ __launch_bounds__(256)
void gemm_bf16_k(const __bf16* __restrict__ A, int lda,
                 const __bf16* __restrict__ BT, int ldbt,
                 float* __restrict__ C, int ldc,
                 int M, int N, int K, int act)
{
    __shared__ __bf16 As[2][256*32];
    __shared__ __bf16 Bs[2][64*32];    // [col][k], k contiguous

    const int tid  = threadIdx.x;
    const int lane = tid & 31;
    const int wv   = tid >> 5;                    // 0..7 : 32-row stripe
    const int rowBase = blockIdx.y * 256;
    const int colBase = blockIdx.x * 64;

    const int bcol = tid >> 2, bk = (tid & 3) * 8;      // B: 64 cols x 32 k
    const __bf16* aP = A  + (size_t)(rowBase + tid)  * lda;           // A: row per thread
    const __bf16* bP = BT + (size_t)(colBase + bcol) * ldbt + bk;

    v8f acc[2][4] = {};
    const int nk = K >> 5;

    // prologue: async-copy tile 0 into buffer 0
    #pragma unroll
    for (int j = 0; j < 4; ++j)
        async_copy_b128(aP + j*8, &As[0][tid*32 + j*8]);
    async_copy_b128(bP, &Bs[0][bcol*32 + bk]);

    const int m     = lane & 15;
    const int half  = lane >> 4;
    const int aoff0 = (wv*32      + m)*32 + half*8;   // A frag: K = half*8+e | 16+half*8+e
    const int aoff1 = (wv*32 + 16 + m)*32 + half*8;

    for (int kt = 0; kt < nk; ++kt) {
        const int cur = kt & 1;
        if (kt + 1 < nk) {                        // issue next tile's DMA, keep 5 in flight
            const __bf16* ap = aP + (size_t)(kt+1) * 32;
            const __bf16* bp = bP + (size_t)(kt+1) * 32;
            #pragma unroll
            for (int j = 0; j < 4; ++j)
                async_copy_b128(ap + j*8, &As[cur^1][tid*32 + j*8]);
            async_copy_b128(bp, &Bs[cur^1][bcol*32 + bk]);
            asm volatile("s_wait_asynccnt 0x5" ::: "memory");
        } else {
            asm volatile("s_wait_asynccnt 0x0" ::: "memory");
        }
        if (kt + 2 < nk) {                        // L2 prefetch two tiles ahead
            __builtin_prefetch(aP + (size_t)(kt+2) * 32, 0, 1);
            __builtin_prefetch(bP + (size_t)(kt+2) * 32, 0, 1);
        }
        __syncthreads();                          // buf[cur] complete on all waves

        // hoist ALL fragment loads before the WMMA block (partial dscnt waits)
        v8bf a0l = *(const v8bf*)&As[cur][aoff0];
        v8bf a0h = *(const v8bf*)&As[cur][aoff0 + 16];
        v8bf a1l = *(const v8bf*)&As[cur][aoff1];
        v8bf a1h = *(const v8bf*)&As[cur][aoff1 + 16];
        v8bf brl[4], brh[4];
        #pragma unroll
        for (int jt = 0; jt < 4; ++jt) {          // B frag: lane n=m, K = half*8+e | 16+half*8+e
            const int boff = (jt*16 + m)*32 + half*8;
            brl[jt] = *(const v8bf*)&Bs[cur][boff];
            brh[jt] = *(const v8bf*)&Bs[cur][boff + 16];
        }
        v16bf af0, af1;
        #pragma unroll
        for (int j = 0; j < 8; ++j) {
            af0[j] = a0l[j]; af0[j+8] = a0h[j];
            af1[j] = a1l[j]; af1[j+8] = a1h[j];
        }
        #pragma unroll
        for (int jt = 0; jt < 4; ++jt) {
            v16bf bf;
            #pragma unroll
            for (int j = 0; j < 8; ++j) { bf[j] = brl[jt][j]; bf[j+8] = brh[jt][j]; }
            acc[0][jt] = __builtin_amdgcn_wmma_f32_16x16x32_bf16(false, af0, false, bf,
                                                                 (short)0, acc[0][jt], false, false);
            acc[1][jt] = __builtin_amdgcn_wmma_f32_16x16x32_bf16(false, af1, false, bf,
                                                                 (short)0, acc[1][jt], false, false);
        }
        __syncthreads();                          // done reading buf[cur]
    }

    // C layout: VGPR r -> M = r + 8*half, N = lane%16
    #pragma unroll
    for (int i = 0; i < 2; ++i) {
        #pragma unroll
        for (int jt = 0; jt < 4; ++jt) {
            #pragma unroll
            for (int r = 0; r < 8; ++r) {
                int row = rowBase + wv*32 + i*16 + r + 8*half;
                int col = colBase + jt*16 + m;
                float xv = acc[i][jt][r];
                if (act == 1) xv = sigmoidf_(xv);
                C[(size_t)row*ldc + col] = xv;
            }
        }
    }
}

// ---------------------------------------------------------------- pointwise
__global__ __launch_bounds__(128)
void pointwise1_k(const float* __restrict__ Ycat, const float* __restrict__ pdr,
                  float* __restrict__ q, float* __restrict__ k, float* __restrict__ v,
                  __bf16* __restrict__ hb, __bf16* __restrict__ hdb)
{
    __shared__ float red[128];
    const int t   = blockIdx.x;
    const int tid = threadIdx.x;
    const float* y = Ycat + (size_t)t * NCAT;

    const int grp = tid >> 6;   // 0: q, 1: k
    const int d   = tid & 63;
    float pre = (grp == 0) ? y[d] : y[64 + d];
    red[tid] = pre * pre;
    __syncthreads();
    #pragma unroll
    for (int s = 32; s > 0; s >>= 1) { if (d < s) red[tid] += red[tid + s]; __syncthreads(); }
    float nrm = sqrtf(red[grp * 64]);
    float xn  = pre / fmaxf(nrm, 1e-12f);
    float mu  = (xn > 20.0f) ? xn : log1pf(expf(xn));                  // softplus
    float freq = powf(10000.0f, (float)d * (1.0f/64.0f));
    float shift = (grp == 0) ? 0.0f : -6.283185307179586f * sigmoidf_(pdr[d]);
    float phi = (float)t * freq + shift;
    float* dst = (grp == 0) ? q : k;
    dst[(size_t)t*128 + d]      = mu * cosf(phi);
    dst[(size_t)t*128 + 64 + d] = mu * sinf(phi);

    if (tid < 64) {
        float yv = y[128 + tid];
        v[(size_t)t*64 + tid] = siluf_(yv);
    }
    {   // alpha hidden: silu(ag)*au, padded 103->128
        int c = tid; float h = 0.0f;
        if (c < 103) { float g = y[192 + c], u = y[304 + c]; h = siluf_(g) * u; }
        hb[(size_t)t*128 + c] = (__bf16)h;
    }
    if (tid < 64) {   // gate hidden: silu(x@W_d), padded 39->64
        int c = tid; float h = 0.0f;
        if (c < 39) h = siluf_(y[416 + c]);
        hdb[(size_t)t*64 + c] = (__bf16)h;
    }
}

// ---------------------------------------------------------------- sequential scan
// Columns of S are independent: S_new[:,c] = a.S[:,c] + k*(v_c - k.(a.S[:,c])).
// One wave per column (64 waves total), zero barriers; state 128 f32 = 4 regs/lane.
// out[t,c] = q.(a.S[:,c]) + (q.k)*(v_c - w).
__global__ __launch_bounds__(256)
void scan_col_k(const float* __restrict__ alpha, const float* __restrict__ kmat,
                const float* __restrict__ vmat, const float* __restrict__ qmat,
                float* __restrict__ sout)
{
    const int lane = threadIdx.x & 31;
    const int wv   = threadIdx.x >> 5;
    const int c    = blockIdx.x * 8 + wv;        // 0..63
    const int r4   = lane * 4;

    float4 s = make_float4(0.f, 0.f, 0.f, 0.f);

    for (int t = 0; t < T_SEQ; ++t) {
        const float4 av = *(const float4*)(alpha + (size_t)t*128 + r4);
        const float4 kv = *(const float4*)(kmat  + (size_t)t*128 + r4);
        const float4 qv = *(const float4*)(qmat  + (size_t)t*128 + r4);

        s.x *= av.x; s.y *= av.y; s.z *= av.z; s.w *= av.w;

        float kw = kv.x*s.x + kv.y*s.y + kv.z*s.z + kv.w*s.w;
        float qa = qv.x*s.x + qv.y*s.y + qv.z*s.z + qv.w*s.w;
        float qk = qv.x*kv.x + qv.y*kv.y + qv.z*kv.z + qv.w*kv.w;
        #pragma unroll
        for (int off = 16; off > 0; off >>= 1) {
            kw += __shfl_xor(kw, off, 32);
            qa += __shfl_xor(qa, off, 32);
            qk += __shfl_xor(qk, off, 32);
        }

        float vc = vmat[(size_t)t*64 + c];
        float d  = vc - kw;
        s.x += kv.x*d; s.y += kv.y*d; s.z += kv.z*d; s.w += kv.w*d;

        if (lane == 0) sout[(size_t)t*64 + c] = qa + qk*d;
    }
}

// ---------------------------------------------------------------- RMSNorm * gate
__global__ __launch_bounds__(64)
void rmsgate_k(const float* __restrict__ sout, const float* __restrict__ gnorm,
               const float* __restrict__ gate, __bf16* __restrict__ outb)
{
    __shared__ float red[64];
    const int t = blockIdx.x, c = threadIdx.x;
    float s = sout[(size_t)t*64 + c];
    red[c] = s * s;
    __syncthreads();
    #pragma unroll
    for (int k = 32; k > 0; k >>= 1) { if (c < k) red[c] += red[c + k]; __syncthreads(); }
    float r = rsqrtf(red[0] * (1.0f/64.0f) + 1.1920928955078125e-7f);
    float o = s * r * gnorm[c] * gate[(size_t)t*64 + c];
    outb[(size_t)t*64 + c] = (__bf16)o;
}

// ---------------------------------------------------------------- launch
extern "C" void kernel_launch(void* const* d_in, const int* in_sizes, int n_in,
                              void* d_out, int out_size, void* d_ws, size_t ws_size,
                              hipStream_t stream)
{
    const float* x    = (const float*)d_in[0];
    const float* Wq   = (const float*)d_in[1];
    const float* Wk   = (const float*)d_in[2];
    const float* Wv   = (const float*)d_in[3];
    const float* pdr  = (const float*)d_in[4];
    const float* Wag  = (const float*)d_in[5];
    const float* Wau  = (const float*)d_in[6];
    const float* Wad  = (const float*)d_in[7];
    const float* gnm  = (const float*)d_in[8];
    const float* Wd   = (const float*)d_in[9];
    const float* Wu   = (const float*)d_in[10];
    const float* Wout = (const float*)d_in[11];
    float* out = (float*)d_out;

    char* ws = (char*)d_ws;
    size_t off = 0;
    auto alloc = [&](size_t bytes) { size_t o = off; off += (bytes + 255) & ~(size_t)255; return o; };

    __bf16* xb    = (__bf16*)(ws + alloc((size_t)T_SEQ*D_IN*2));
    __bf16* WcatT = (__bf16*)(ws + alloc((size_t)NCAT*D_IN*2));
    __bf16* WadT  = (__bf16*)(ws + alloc((size_t)128*128*2));
    __bf16* WuT   = (__bf16*)(ws + alloc((size_t)64*64*2));
    __bf16* WoutT = (__bf16*)(ws + alloc((size_t)1024*64*2));
    float*  Ycat  = (float*) (ws + alloc((size_t)T_SEQ*NCAT*4));
    float*  qbuf  = (float*) (ws + alloc((size_t)T_SEQ*128*4));
    float*  kbuf  = (float*) (ws + alloc((size_t)T_SEQ*128*4));
    float*  vbuf  = (float*) (ws + alloc((size_t)T_SEQ*64*4));
    __bf16* hb    = (__bf16*)(ws + alloc((size_t)T_SEQ*128*2));
    __bf16* hdb   = (__bf16*)(ws + alloc((size_t)T_SEQ*64*2));
    float*  abuf  = (float*) (ws + alloc((size_t)T_SEQ*128*4));
    float*  gbuf  = (float*) (ws + alloc((size_t)T_SEQ*64*4));
    float*  sout  = (float*) (ws + alloc((size_t)T_SEQ*64*4));
    __bf16* outb  = (__bf16*)(ws + alloc((size_t)T_SEQ*64*2));
    (void)ws_size; (void)n_in; (void)in_sizes; (void)out_size;

    // 1. pack transposed weights + convert x to bf16
    pack_weights_k<<<256, 256, 0, stream>>>(Wq, Wk, Wv, Wag, Wau, Wd, Wad, Wu, Wout,
                                            WcatT, WadT, WuT, WoutT);
    convert_x_k<<<4096, 256, 0, stream>>>(x, xb, T_SEQ*D_IN);

    // 2. fused projection GEMM: Ycat = x @ Wcat   [8192,512], K=1024
    gemm_bf16_k<<<dim3(NCAT/64, T_SEQ/256), 256, 0, stream>>>(
        xb, D_IN, WcatT, D_IN, Ycat, NCAT, T_SEQ, NCAT, D_IN, 0);

    // 3. pointwise: PoPE(q,k), silu(v), alpha/gate hidden
    pointwise1_k<<<T_SEQ, 128, 0, stream>>>(Ycat, pdr, qbuf, kbuf, vbuf, hb, hdb);

    // 4. alpha = sigmoid(h @ W_ad)   [8192,128], K=128
    gemm_bf16_k<<<dim3(2, T_SEQ/256), 256, 0, stream>>>(
        hb, 128, WadT, 128, abuf, 128, T_SEQ, 128, 128, 1);

    // 5. gate = sigmoid(hd @ W_u)    [8192,64], K=64
    gemm_bf16_k<<<dim3(1, T_SEQ/256), 256, 0, stream>>>(
        hdb, 64, WuT, 64, gbuf, 64, T_SEQ, 64, 64, 1);

    // 6. per-column delta-rule scans (64 independent waves, no barriers)
    scan_col_k<<<8, 256, 0, stream>>>(abuf, kbuf, vbuf, qbuf, sout);

    // 7. RMSNorm * gate -> bf16
    rmsgate_k<<<T_SEQ, 64, 0, stream>>>(sout, gnm, gbuf, outb);

    // 8. final projection: out = outb @ W_out   [8192,1024], K=64
    gemm_bf16_k<<<dim3(1024/64, T_SEQ/256), 256, 0, stream>>>(
        outb, 64, WoutT, 1024, out, 1024, T_SEQ, 1024, 64, 0);
}
